// SNAT_38508676776205
// MI455X (gfx1250) — matrix operand order repaired and checked by reading, not compiled, not run on hardware
//
#include <hip/hip_runtime.h>

#define NN 50000
#define NE 800000
#define IN_DIM 128
#define HD 256          // HEADS*HIDDEN
#define HEADS 4
#define DHEAD 64
#define NEG_SLOPE 0.2f

typedef __attribute__((ext_vector_type(16))) __bf16 v16bf;
typedef __attribute__((ext_vector_type(8)))  float  v8f;

__device__ __forceinline__ unsigned short f32_to_bf16(float f) {
    unsigned int u = __float_as_uint(f);
    // round-to-nearest-even
    u = (u + 0x7FFFu + ((u >> 16) & 1u)) >> 16;
    return (unsigned short)u;
}

// ---------- cast f32 -> bf16 (row-major, same layout) ----------
__global__ void cast_bf16_kernel(const float* __restrict__ in,
                                 unsigned short* __restrict__ out, int n) {
    int tid = blockIdx.x * blockDim.x + threadIdx.x;
    if (tid < n) out[tid] = f32_to_bf16(in[tid]);
}

// ---------- W [K,HD] f32 -> WT [HD,K] bf16 (transpose) ----------
__global__ void prep_wt_kernel(const float* __restrict__ W,
                               unsigned short* __restrict__ WT, int K) {
    int tid = blockIdx.x * blockDim.x + threadIdx.x;
    if (tid >= K * HD) return;
    int row = tid / HD;      // 0..K-1
    int col = tid - row * HD;
    WT[(size_t)col * K + row] = f32_to_bf16(W[tid]);
}

// ---------- zero agg, mord, denom ----------
__global__ void init_kernel(float* __restrict__ agg,
                            unsigned int* __restrict__ mord,
                            float* __restrict__ denom) {
    int tid = blockIdx.x * blockDim.x + threadIdx.x;
    if (tid < NN * HD) agg[tid] = 0.0f;
    if (tid < NN * HEADS) { mord[tid] = 0u; denom[tid] = 0.0f; }
}

// ---------- WMMA GEMM: C[N,HD] = A[N,K](bf16) * B[K,HD] via BT[HD,K](bf16) ----------
// block = 128 threads (4 waves); each wave -> one 16x16 output tile.
// grid = (N/16, HD/16/4)
__global__ void gemm_wmma_kernel(const unsigned short* __restrict__ A,
                                 const unsigned short* __restrict__ BT,
                                 float* __restrict__ C, int K) {
    const int wave = threadIdx.x >> 5;
    const int lane = threadIdx.x & 31;
    const int mtile = blockIdx.x;
    const int ntile = blockIdx.y * 4 + wave;

    const int m   = lane & 15;            // A row within tile, also B column
    const int kbA = (lane >> 4) << 3;     // 0 or 8  (A half-K interleave)
    const int kbB = (lane >> 4) << 3;     // 0 or 8  (B half-K interleave)

    const unsigned short* arow = A  + (size_t)(mtile * 16 + m) * K + kbA;
    const unsigned short* brow = BT + (size_t)(ntile * 16 + m) * K + kbB;

    v8f acc = {};
    for (int k0 = 0; k0 < K; k0 += 32) {
        union { uint4 u[2]; v16bf v; } ua, ub;
        // A: K runs [kbA, kbA+8) and [kbA+16, kbA+24)
        ua.u[0] = *(const uint4*)(arow + k0);
        ua.u[1] = *(const uint4*)(arow + k0 + 16);
        // B: K runs [kbB, kbB+8) and [kbB+16, kbB+24)
        ub.u[0] = *(const uint4*)(brow + k0);
        ub.u[1] = *(const uint4*)(brow + k0 + 16);
        acc = __builtin_amdgcn_wmma_f32_16x16x32_bf16(
                  false, ua.v, false, ub.v, (short)0, acc, false, false);
    }

    const int col   = ntile * 16 + (lane & 15);
    const int rbase = mtile * 16 + ((lane >> 4) << 3);
#pragma unroll
    for (int r = 0; r < 8; ++r)
        C[(size_t)(rbase + r) * HD + col] = acc[r];
}

// ---------- el/er: per (node, head) dot of h with al/ar ----------
__global__ void elr_kernel(const float* __restrict__ h,
                           const float* __restrict__ al, const float* __restrict__ ar,
                           float* __restrict__ el, float* __restrict__ er) {
    int tid = blockIdx.x * blockDim.x + threadIdx.x;
    if (tid >= NN * HEADS) return;
    int n = tid >> 2, hh = tid & 3;
    const float4* hp  = (const float4*)(h + (size_t)n * HD + hh * DHEAD);
    const float4* alp = (const float4*)(al + hh * DHEAD);
    const float4* arp = (const float4*)(ar + hh * DHEAD);
    float sl = 0.f, sr = 0.f;
#pragma unroll 4
    for (int i = 0; i < DHEAD / 4; ++i) {
        float4 hv = hp[i], a = alp[i], b = arp[i];
        sl += hv.x * a.x + hv.y * a.y + hv.z * a.z + hv.w * a.w;
        sr += hv.x * b.x + hv.y * b.y + hv.z * b.z + hv.w * b.w;
    }
    el[tid] = sl; er[tid] = sr;
}

// ---------- edge pass 1: e = leakyrelu(el[src]+er[dst]); segment-max(dst) ----------
__global__ void edge1_kernel(const int* __restrict__ src, const int* __restrict__ dst,
                             const float* __restrict__ el, const float* __restrict__ er,
                             float* __restrict__ eedge, unsigned int* __restrict__ mord) {
    int tid = blockIdx.x * blockDim.x + threadIdx.x;
    if (tid >= NE * HEADS) return;
    int e = tid >> 2, hh = tid & 3;
    int s = src[e], d = dst[e];
    float v = el[s * HEADS + hh] + er[d * HEADS + hh];
    v = v > 0.f ? v : NEG_SLOPE * v;
    eedge[tid] = v;
    unsigned int u = __float_as_uint(v);
    unsigned int ord = ((int)u >= 0) ? (u | 0x80000000u) : ~u;   // order-preserving
    atomicMax(&mord[d * HEADS + hh], ord);
}

// ---------- decode ordered-uint max back to float (0 if no edges) ----------
__global__ void decode_m_kernel(unsigned int* __restrict__ mord) {
    int tid = blockIdx.x * blockDim.x + threadIdx.x;
    if (tid >= NN * HEADS) return;
    unsigned int u = mord[tid];
    float f;
    if (u == 0u)                f = 0.0f;
    else if (u & 0x80000000u)   f = __uint_as_float(u ^ 0x80000000u);
    else                        f = __uint_as_float(~u);
    ((float*)mord)[tid] = f;
}

// ---------- edge pass 2: ex = exp(e - m[dst]); segment-sum(dst) ----------
__global__ void edge2_kernel(const int* __restrict__ dst,
                             float* __restrict__ eedge,
                             const float* __restrict__ m,
                             float* __restrict__ denom) {
    int tid = blockIdx.x * blockDim.x + threadIdx.x;
    if (tid >= NE * HEADS) return;
    int e = tid >> 2, hh = tid & 3;
    int d = dst[e];
    float ex = __expf(eedge[tid] - m[d * HEADS + hh]);
    eedge[tid] = ex;
    atomicAdd(&denom[d * HEADS + hh], ex);
}

// ---------- edge pass 3 (dominant traffic): agg[dst] += h[src] * alpha ----------
// one wave per edge; lane handles 8 contiguous floats of the 256-float row
__global__ void edge3_kernel(const int* __restrict__ src, const int* __restrict__ dst,
                             const float* __restrict__ h,
                             const float* __restrict__ eedge,
                             const float* __restrict__ denom,
                             float* __restrict__ agg) {
    int eidx = blockIdx.x * (blockDim.x >> 5) + (threadIdx.x >> 5);
    if (eidx >= NE) return;
    int lane = threadIdx.x & 31;
    int s = src[eidx], d = dst[eidx];
    int head = lane >> 3;                 // 8 floats per lane => 8 lanes per head
    float alpha = eedge[eidx * HEADS + head] /
                  (denom[d * HEADS + head] + 1e-9f);
    const float4* hp = (const float4*)(h + (size_t)s * HD + lane * 8);
    float4 a = hp[0], b = hp[1];
    float* op = agg + (size_t)d * HD + lane * 8;
    atomicAdd(op + 0, a.x * alpha);
    atomicAdd(op + 1, a.y * alpha);
    atomicAdd(op + 2, a.z * alpha);
    atomicAdd(op + 3, a.w * alpha);
    atomicAdd(op + 4, b.x * alpha);
    atomicAdd(op + 5, b.y * alpha);
    atomicAdd(op + 6, b.z * alpha);
    atomicAdd(op + 7, b.w * alpha);
}

// ---------- ELU + cast to bf16 for next layer's GEMM input ----------
__global__ void elu_cast_kernel(const float* __restrict__ agg,
                                unsigned short* __restrict__ xb) {
    int tid = blockIdx.x * blockDim.x + threadIdx.x;
    if (tid >= NN * HD) return;
    float v = agg[tid];
    v = v > 0.f ? v : (__expf(v) - 1.f);
    xb[tid] = f32_to_bf16(v);
}

// ---------- final: elu, head-mean, @Wout + bout, relu; wave per node ----------
__global__ void final_kernel(const float* __restrict__ agg,
                             const float* __restrict__ Wout,
                             const float* __restrict__ bout,
                             float* __restrict__ out) {
    int node = blockIdx.x * (blockDim.x >> 5) + (threadIdx.x >> 5);
    if (node >= NN) return;
    int lane = threadIdx.x & 31;
    float partial = 0.f;
#pragma unroll
    for (int d = lane; d < DHEAD; d += 32) {
        float s = 0.f;
#pragma unroll
        for (int hh = 0; hh < HEADS; ++hh) {
            float v = agg[(size_t)node * HD + hh * DHEAD + d];
            v = v > 0.f ? v : (__expf(v) - 1.f);
            s += v;
        }
        partial += (s * (1.0f / HEADS)) * Wout[d];
    }
#pragma unroll
    for (int off = 16; off > 0; off >>= 1)
        partial += __shfl_down(partial, off, 32);
    if (lane == 0) out[node] = fmaxf(partial + bout[0], 0.f);
}

extern "C" void kernel_launch(void* const* d_in, const int* in_sizes, int n_in,
                              void* d_out, int out_size, void* d_ws, size_t ws_size,
                              hipStream_t stream) {
    const float* x    = (const float*)d_in[0];
    const int*   src  = (const int*)d_in[1];
    const int*   dst  = (const int*)d_in[2];
    const float* W[3]  = {(const float*)d_in[3], (const float*)d_in[6], (const float*)d_in[9]};
    const float* al[3] = {(const float*)d_in[4], (const float*)d_in[7], (const float*)d_in[10]};
    const float* ar[3] = {(const float*)d_in[5], (const float*)d_in[8], (const float*)d_in[11]};
    const float* Wout = (const float*)d_in[12];
    const float* bout = (const float*)d_in[13];
    float* out = (float*)d_out;

    // workspace carve-up (256B aligned)
    size_t off = 0;
    auto carve = [&](size_t bytes) -> char* {
        char* p = (char*)d_ws + off;
        off += (bytes + 255) & ~(size_t)255;
        return p;
    };
    unsigned short* xb   = (unsigned short*)carve((size_t)NN * HD * 2);     // bf16 layer input
    unsigned short* wt   = (unsigned short*)carve((size_t)HD * HD * 2);     // bf16 W^T
    float*          hbuf = (float*)carve((size_t)NN * HD * 4);              // h = x@W
    float*          elb  = (float*)carve((size_t)NN * HEADS * 4);
    float*          erb  = (float*)carve((size_t)NN * HEADS * 4);
    unsigned int*   mord = (unsigned int*)carve((size_t)NN * HEADS * 4);    // max (ordered), then float
    float*          den  = (float*)carve((size_t)NN * HEADS * 4);
    float*          eedg = (float*)carve((size_t)NE * HEADS * 4);           // e, then ex
    float*          agg  = (float*)carve((size_t)NN * HD * 4);

    // layer 0 input cast
    cast_bf16_kernel<<<(NN * IN_DIM + 255) / 256, 256, 0, stream>>>(x, xb, NN * IN_DIM);

    for (int l = 0; l < 3; ++l) {
        const int K = (l == 0) ? IN_DIM : HD;

        prep_wt_kernel<<<(K * HD + 255) / 256, 256, 0, stream>>>(W[l], wt, K);
        init_kernel<<<(NN * HD + 255) / 256, 256, 0, stream>>>(agg, mord, den);

        gemm_wmma_kernel<<<dim3(NN / 16, (HD / 16) / 4), 128, 0, stream>>>(xb, wt, hbuf, K);

        elr_kernel<<<(NN * HEADS + 255) / 256, 256, 0, stream>>>(hbuf, al[l], ar[l], elb, erb);

        edge1_kernel<<<(NE * HEADS + 255) / 256, 256, 0, stream>>>(src, dst, elb, erb, eedg, mord);
        decode_m_kernel<<<(NN * HEADS + 255) / 256, 256, 0, stream>>>(mord);
        edge2_kernel<<<(NE * HEADS + 255) / 256, 256, 0, stream>>>(dst, eedg, (const float*)mord, den);
        edge3_kernel<<<(NE + 7) / 8, 256, 0, stream>>>(src, dst, hbuf, eedg, den, agg);

        if (l < 2)
            elu_cast_kernel<<<(NN * HD + 255) / 256, 256, 0, stream>>>(agg, xb);
    }

    final_kernel<<<(NN + 7) / 8, 256, 0, stream>>>(agg, Wout, bout, out);
}
